// Seq2SeqVAE_54150947668723
// MI455X (gfx1250) — compile-verified, matrix operand order checked by reading, not run on hardware
//
#include <hip/hip_runtime.h>
#include <hip/hip_bf16.h>

typedef __bf16 bf16;
typedef __attribute__((ext_vector_type(8)))  __bf16 v8bf;
typedef __attribute__((ext_vector_type(16))) __bf16 v16bf;
typedef __attribute__((ext_vector_type(8)))  float  v8f;
typedef __attribute__((ext_vector_type(4)))  float  v4f;
typedef __attribute__((ext_vector_type(4)))  int    v4i;

#define HID   512
#define LAT   128
#define OUTD  64
#define INPD  64
#define BATCH 512
#define DKIN  192   /* OUTD + LAT */

#define BROW  40    /* padded LDS row stride (elements) for a 32-K tile row */
#define BTILE 128   /* rows per B tile: 4 gates x 32 hidden cols */

#define GLOBAL_AS __attribute__((address_space(1)))
#define LDS_AS    __attribute__((address_space(3)))

// ---------------- device helpers ----------------

__device__ __forceinline__ float fsig(float x)  { return 1.0f / (1.0f + __expf(-x)); }
__device__ __forceinline__ float ftanh(float x) { float e = __expf(2.0f * x); return 1.0f - 2.0f / (e + 1.0f); }

__device__ __forceinline__ v8f wmma_bf16(v16bf a, v16bf b, v8f c) {
    return __builtin_amdgcn_wmma_f32_16x16x32_bf16(false, a, false, b, (short)0, c, false, false);
}

// 16-byte global -> LDS async copy (per lane), ASYNCcnt-tracked.
__device__ __forceinline__ void async_cp16(const bf16* g, bf16* l) {
#if defined(__has_builtin) && __has_builtin(__builtin_amdgcn_global_load_async_to_lds_b128)
    __builtin_amdgcn_global_load_async_to_lds_b128(
        (GLOBAL_AS v4i*)(v4i*)const_cast<bf16*>(g),
        (LDS_AS v4i*)(v4i*)l, 0, 0);
#else
    *(v8bf*)l = *(const v8bf*)g;   // fallback: sync copy via VGPRs (ds_store_b128)
#endif
}

__device__ __forceinline__ void wait_async_le2() {
#if defined(__has_builtin) && __has_builtin(__builtin_amdgcn_s_wait_asynccnt)
    __builtin_amdgcn_s_wait_asynccnt(2);
#else
    asm volatile("s_wait_asynccnt 0x2" ::: "memory");
#endif
}
__device__ __forceinline__ void wait_async_0() {
#if defined(__has_builtin) && __has_builtin(__builtin_amdgcn_s_wait_asynccnt)
    __builtin_amdgcn_s_wait_asynccnt(0);
#else
    asm volatile("s_wait_asynccnt 0x0" ::: "memory");
#endif
}

// Load a 16x32 bf16 WMMA operand fragment from row-major [row][K] global storage.
// lane l<16 : row row0+l,    K = k0+{0..7, 16..23}
// lane l>=16: row row0+l-16, K = k0+{8..15, 24..31}
__device__ __forceinline__ v16bf frag_bf16(const bf16* __restrict__ base, int stride, int row0, int k0) {
    int lane = threadIdx.x & 31;
    int lh   = lane & 15;
    int ks   = (lane >> 4) << 3;
    const bf16* p = base + (size_t)(row0 + lh) * stride + k0 + ks;
    v8bf lo = *(const v8bf*)(p);
    v8bf hi = *(const v8bf*)(p + 16);
    v16bf r;
#pragma unroll
    for (int i = 0; i < 8; ++i) { r[i] = lo[i]; r[i + 8] = hi[i]; }
    return r;
}

// Same fragment but sourced from fp32 (converted on the fly) — encoder x.
__device__ __forceinline__ v16bf frag_f32(const float* __restrict__ base, int stride, int row0, int k0) {
    int lane = threadIdx.x & 31;
    int lh   = lane & 15;
    int ks   = (lane >> 4) << 3;
    const float* p = base + (size_t)(row0 + lh) * stride + k0 + ks;
    v4f a0 = *(const v4f*)(p);
    v4f a1 = *(const v4f*)(p + 4);
    v4f b0 = *(const v4f*)(p + 16);
    v4f b1 = *(const v4f*)(p + 20);
    v16bf r;
#pragma unroll
    for (int i = 0; i < 4; ++i) {
        r[i]      = (bf16)a0[i];
        r[i + 4]  = (bf16)a1[i];
        r[i + 8]  = (bf16)b0[i];
        r[i + 12] = (bf16)b1[i];
    }
    return r;
}

// B fragment from an LDS tile: rows stored with BROW-element pitch, 32 K per row.
__device__ __forceinline__ v16bf frag_lds(const bf16* buf, int row0) {
    int lane = threadIdx.x & 31;
    int lh   = lane & 15;
    int ks   = (lane >> 4) << 3;
    const bf16* p = buf + (row0 + lh) * BROW + ks;
    v8bf lo = *(const v8bf*)(p);
    v8bf hi = *(const v8bf*)(p + 16);
    v16bf r;
#pragma unroll
    for (int i = 0; i < 8; ++i) { r[i] = lo[i]; r[i + 8] = hi[i]; }
    return r;
}

// Stage one 128-row x 32-K B tile (4 gates x 32 hidden cols) into LDS.
// 256 threads x 2 iters x 16B = 8KB; 2 async ops per thread.
__device__ __forceinline__ void prefetch_B(const bf16* __restrict__ W, int Kdim,
                                           int c0base, int kk, bf16* buf) {
    const int tid = threadIdx.x;
    const int seg = tid & 3;
#pragma unroll
    for (int j = 0; j < 2; ++j) {
        const int lr  = (tid >> 2) + j * 64;        // 0..127
        const int g   = lr >> 5;
        const int col = lr & 31;
        const bf16* gsrc = W + (size_t)(g * HID + c0base + col) * Kdim + kk + seg * 8;
        bf16* ldst = buf + lr * BROW + seg * 8;
        async_cp16(gsrc, ldst);
    }
}

// ---------------- kernels ----------------

__global__ void cvt_f32_bf16_k(const float* __restrict__ src, bf16* __restrict__ dst, int n) {
    for (int i = blockIdx.x * blockDim.x + threadIdx.x; i < n; i += gridDim.x * blockDim.x)
        dst[i] = (bf16)src[i];
}

__global__ void zero_bf16_k(bf16* __restrict__ p, int n) {
    for (int i = blockIdx.x * blockDim.x + threadIdx.x; i < n; i += gridDim.x * blockDim.x)
        p[i] = (bf16)0.0f;
}

__global__ void zero_f32_k(float* __restrict__ p, int n) {
    for (int i = blockIdx.x * blockDim.x + threadIdx.x; i < n; i += gridDim.x * blockDim.x)
        p[i] = 0.0f;
}

// One LSTM time step.  grid = (BATCH/64, HID/32), block = 256 (8 waves).
// WG tile: 64 batch rows x 32 hidden cols; wave w -> M tile (w&3), col group (w>>2).
// Weights double-buffered through LDS via async global->LDS copies; each B tile
// is fetched once per WG and consumed by 4 waves.
__global__ void __launch_bounds__(256)
lstm_step_kernel(const float* __restrict__ xin_f32,  // fp32 input rows (encoder) or null
                 const bf16*  __restrict__ xin_bf,   // bf16 input rows (decoder) or null
                 int x_stride, int k_in,
                 const bf16*  __restrict__ Wih,      // [4H][k_in] bf16
                 const bf16*  __restrict__ Whh,      // [4H][H]    bf16
                 const float* __restrict__ bias,     // [4H]       f32 (b_ih+b_hh)
                 const bf16*  __restrict__ h_in,     // [B][H] bf16 (prev step)
                 bf16*        __restrict__ h_out,    // [B][H] bf16 (this step)
                 float*       __restrict__ c)        // [B][H] f32 (in/out)
{
    __shared__ bf16 ldsB[2][BTILE * BROW];

    const int w      = threadIdx.x >> 5;
    const int mt     = w & 3;
    const int ct     = w >> 2;
    const int m0     = blockIdx.x * 64 + mt * 16;      // batch tile
    const int c0base = blockIdx.y * 32;                // WG hidden-col base
    const int c0     = c0base + ct * 16;               // wave hidden-col tile

    const int nIn  = k_in >> 5;          // Wih chunks (2 enc / 6 dec)
    const int nTot = nIn + (HID >> 5);   // + 16 Whh chunks

    v8f acc[4];
#pragma unroll
    for (int g = 0; g < 4; ++g) acc[g] = (v8f){0.f, 0.f, 0.f, 0.f, 0.f, 0.f, 0.f, 0.f};

    // prime the double buffer with chunks 0 and 1 (always Wih chunks: nIn >= 2)
    prefetch_B(Wih, k_in, c0base, 0,  &ldsB[0][0]);
    prefetch_B(Wih, k_in, c0base, 32, &ldsB[1][0]);

    for (int s = 0; s < nTot; ++s) {
        // A fragment for this chunk (issue loads before waiting on async copies)
        v16bf a;
        if (s < nIn) {
            const int kk = s * 32;
            a = xin_f32 ? frag_f32(xin_f32, x_stride, m0, kk)
                        : frag_bf16(xin_bf, x_stride, m0, kk);
        } else {
            a = frag_bf16(h_in, HID, m0, (s - nIn) * 32);
        }

        // chunk s complete (in-order completion; chunk s+1 may remain in flight)
        if (s + 1 < nTot) wait_async_le2(); else wait_async_0();
        __syncthreads();

        const bf16* buf = &ldsB[s & 1][0];
#pragma unroll
        for (int g = 0; g < 4; ++g)
            acc[g] = wmma_bf16(a, frag_lds(buf, g * 32 + ct * 16), acc[g]);

        __syncthreads();   // all waves done reading buf before it is refilled

        const int sn = s + 2;
        if (sn < nTot) {
            if (sn < nIn) prefetch_B(Wih, k_in, c0base, sn * 32,          &ldsB[sn & 1][0]);
            else          prefetch_B(Whh, HID,  c0base, (sn - nIn) * 32,  &ldsB[sn & 1][0]);
        }
    }

    // ---- fused cell update (fp32) ----
    const int lane = threadIdx.x & 31;
    const int n    = c0 + (lane & 15);             // hidden column
    const int mb   = m0 + (lane >> 4) * 8;         // batch row base
    const float bi  = bias[0 * HID + n];
    const float bf_ = bias[1 * HID + n];
    const float bg  = bias[2 * HID + n];
    const float bo  = bias[3 * HID + n];

#pragma unroll
    for (int r = 0; r < 8; ++r) {
        const int row = mb + r;
        const size_t idx = (size_t)row * HID + n;
        const float iv = fsig (acc[0][r] + bi);
        const float fv = fsig (acc[1][r] + bf_);
        const float gv = ftanh(acc[2][r] + bg);
        const float ov = fsig (acc[3][r] + bo);
        const float cn = fv * c[idx] + iv * gv;
        c[idx] = cn;
        h_out[idx] = (bf16)(ov * ftanh(cn));
    }
}

// Decoder output projection: out = h @ out_W^T + out_b.
// grid = BATCH/32, block = 256 (8 waves -> 32 batch x 64 out cols per WG).
__global__ void __launch_bounds__(256)
out_proj_kernel(const bf16* __restrict__ h,      // [B][H] bf16
                const bf16* __restrict__ Wo,     // [64][H] bf16
                const float* __restrict__ bo,    // [64]
                float* __restrict__ recon,       // [B][tgt][64]
                int t, int tgt_len,
                bf16* __restrict__ dec_in)       // [B][DKIN], cols 0..63 = prev
{
    const int w  = threadIdx.x >> 5;
    const int m0 = blockIdx.x * 32 + (w & 1) * 16;
    const int n0 = (w >> 1) * 16;

    v8f acc = (v8f){0.f, 0.f, 0.f, 0.f, 0.f, 0.f, 0.f, 0.f};
    for (int kk = 0; kk < HID; kk += 32) {
        v16bf a = frag_bf16(h,  HID, m0, kk);
        v16bf b = frag_bf16(Wo, HID, n0, kk);
        acc = wmma_bf16(a, b, acc);
    }

    const int lane = threadIdx.x & 31;
    const int n  = n0 + (lane & 15);
    const int mb = m0 + (lane >> 4) * 8;
    const float bb = bo[n];
#pragma unroll
    for (int r = 0; r < 8; ++r) {
        const int row = mb + r;
        const float v = acc[r] + bb;
        recon[((size_t)row * tgt_len + t) * OUTD + n] = v;
        dec_in[(size_t)row * DKIN + n] = (bf16)v;
    }
}

// mu / logvar projections + reparameterization.  One thread per (b, l).
__global__ void finalize_enc_kernel(const bf16* __restrict__ hT,
                                    const float* __restrict__ muW, const float* __restrict__ mub,
                                    const float* __restrict__ lvW, const float* __restrict__ lvb,
                                    const float* __restrict__ eps,
                                    float* __restrict__ mu_out, float* __restrict__ lv_out,
                                    bf16* __restrict__ dec_in)   // z at cols 64..191
{
    const int idx = blockIdx.x * blockDim.x + threadIdx.x;   // B*LAT threads
    const int b = idx >> 7;
    const int l = idx & 127;
    const bf16*  hr = hT  + (size_t)b * HID;
    const float* mw = muW + (size_t)l * HID;
    const float* lw = lvW + (size_t)l * HID;
    float mu = 0.f, lv = 0.f;
    for (int k = 0; k < HID; ++k) {
        const float hv = (float)hr[k];
        mu += hv * mw[k];
        lv += hv * lw[k];
    }
    mu += mub[l];
    lv += lvb[l];
    mu_out[idx] = mu;
    lv_out[idx] = lv;
    const float z = mu + eps[idx] * __expf(0.5f * lv);
    dec_in[(size_t)b * DKIN + 64 + l] = (bf16)z;
}

// ---------------- host ----------------

extern "C" void kernel_launch(void* const* d_in, const int* in_sizes, int n_in,
                              void* d_out, int out_size, void* d_ws, size_t ws_size,
                              hipStream_t stream) {
    const float* x       = (const float*)d_in[0];
    const float* eps     = (const float*)d_in[1];
    const float* enc_Wih = (const float*)d_in[3];
    const float* enc_Whh = (const float*)d_in[4];
    const float* enc_b   = (const float*)d_in[5];
    const float* mu_W    = (const float*)d_in[6];
    const float* mu_b    = (const float*)d_in[7];
    const float* lv_W    = (const float*)d_in[8];
    const float* lv_b    = (const float*)d_in[9];
    const float* dec_Wih = (const float*)d_in[10];
    const float* dec_Whh = (const float*)d_in[11];
    const float* dec_b   = (const float*)d_in[12];
    const float* out_W   = (const float*)d_in[13];
    const float* out_b   = (const float*)d_in[14];

    const int T   = in_sizes[0] / (BATCH * INPD);                       // 1024
    const int tgt = (out_size - 2 * BATCH * LAT) / (BATCH * OUTD);      // 512

    float* recon  = (float*)d_out;
    float* mu_out = recon + (size_t)BATCH * tgt * OUTD;
    float* lv_out = mu_out + (size_t)BATCH * LAT;

    // workspace carve-up (all 256B aligned)
    char* ws = (char*)d_ws;
    size_t off = 0;
    auto alloc = [&](size_t bytes) { void* p = ws + off; off += (bytes + 255) & ~(size_t)255; return p; };
    bf16* wWihE = (bf16*)alloc((size_t)4 * HID * INPD * 2);
    bf16* wWhhE = (bf16*)alloc((size_t)4 * HID * HID  * 2);
    bf16* wWihD = (bf16*)alloc((size_t)4 * HID * DKIN * 2);
    bf16* wWhhD = (bf16*)alloc((size_t)4 * HID * HID  * 2);
    bf16* wWo   = (bf16*)alloc((size_t)OUTD * HID * 2);
    bf16* hA    = (bf16*)alloc((size_t)BATCH * HID * 2);
    bf16* hB    = (bf16*)alloc((size_t)BATCH * HID * 2);
    float* cbuf = (float*)alloc((size_t)BATCH * HID * 4);
    bf16* decin = (bf16*)alloc((size_t)BATCH * DKIN * 2);
    (void)ws_size; (void)n_in;

    // 1) convert weights to bf16 (layout preserved: rows = gate/output cols, K contiguous)
    cvt_f32_bf16_k<<<256, 256, 0, stream>>>(enc_Wih, wWihE, 4 * HID * INPD);
    cvt_f32_bf16_k<<<512, 256, 0, stream>>>(enc_Whh, wWhhE, 4 * HID * HID);
    cvt_f32_bf16_k<<<512, 256, 0, stream>>>(dec_Wih, wWihD, 4 * HID * DKIN);
    cvt_f32_bf16_k<<<512, 256, 0, stream>>>(dec_Whh, wWhhD, 4 * HID * HID);
    cvt_f32_bf16_k<<<64,  256, 0, stream>>>(out_W,   wWo,   OUTD * HID);

    // 2) zero encoder state
    zero_bf16_k<<<256, 256, 0, stream>>>(hA, BATCH * HID);
    zero_f32_k <<<256, 256, 0, stream>>>(cbuf, BATCH * HID);

    const dim3 gridCell(BATCH / 64, HID / 32);

    // 3) encoder: 1024 sequential LSTM steps (x converted on the fly)
    bf16* h_in = hA;
    bf16* h_out = hB;
    for (int t = 0; t < T; ++t) {
        lstm_step_kernel<<<gridCell, 256, 0, stream>>>(
            x + (size_t)t * INPD, nullptr, T * INPD, INPD,
            wWihE, wWhhE, enc_b, h_in, h_out, cbuf);
        bf16* tmp = h_in; h_in = h_out; h_out = tmp;
    }
    bf16* hFinal = h_in;   // last written buffer

    // 4) mu / logvar / z  (z -> decoder input cols 64..191; prev cols zeroed first)
    zero_bf16_k<<<128, 256, 0, stream>>>(decin, BATCH * DKIN);
    finalize_enc_kernel<<<(BATCH * LAT) / 256, 256, 0, stream>>>(
        hFinal, mu_W, mu_b, lv_W, lv_b, eps, mu_out, lv_out, decin);

    // 5) zero decoder state (after finalize has consumed hFinal)
    zero_bf16_k<<<256, 256, 0, stream>>>(hA, BATCH * HID);
    zero_bf16_k<<<256, 256, 0, stream>>>(hB, BATCH * HID);
    zero_f32_k <<<256, 256, 0, stream>>>(cbuf, BATCH * HID);

    // 6) decoder: 512 sequential steps, each = LSTM cell + output projection
    h_in = hA; h_out = hB;
    for (int t = 0; t < tgt; ++t) {
        lstm_step_kernel<<<gridCell, 256, 0, stream>>>(
            nullptr, decin, DKIN, DKIN,
            wWihD, wWhhD, dec_b, h_in, h_out, cbuf);
        out_proj_kernel<<<BATCH / 32, 256, 0, stream>>>(
            h_out, wWo, out_b, recon, t, tgt, decin);
        bf16* tmp = h_in; h_in = h_out; h_out = tmp;
    }
}
